// GedGNN_15487652070014
// MI455X (gfx1250) — matrix operand order repaired and checked by reading, not compiled, not run on hardware
//
#include <hip/hip_runtime.h>
#include <hip/hip_fp16.h>

// ----------------------------------------------------------------------------
// GedGNN / SimGNN forward for MI455X (gfx1250).
//
// Dominant cost: pairwise energy MLP over N^2 pairs. We decompose
//   (af1[i]+af2[j]) @ fc1_W  ==  A1[i] + A2[j]   (A1 = af1@fc1_W, A2 = af2@fc1_W)
// so the heavy stage is a per-pair [64]->[32] GEMM + 32-dot = 9.7 GFLOP,
// executed with v_wmma_f32_16x16x32_f16 (f16 in, f32 accumulate).
// Each wave holds one 16-wide j-tile's A2 slice, packed W2 (B operand) and w3
// in registers and loops over 8 consecutive i values; cross-lane reduction of
// the fc3 dot uses stage-major ds_swizzle xor-butterflies (8 independent rows
// in flight per stage), and a cndmask-select + single predicated store.
//
// Input flattening assumption (setup_inputs dict order, params dict flattened
// recursively in insertion order):
//  0 features_1[N*32] 1 features_2[N*32] 2 hb[1]
//  3 eps1 4 eps2 5 eps3
//  6..11  gin1: W0[32*128] b0[128] W1[128*128] b1[128] gamma[128] beta[128]
// 12..17  gin2: W0[128*64] b0[64]  W1[64*64]  b1[64]  gamma[64]  beta[64]
// 18..23  gin3: W0[64*32]  b0[32]  W1[32*32]  b1[32]  gamma[32]  beta[32]
// 24 fc1_W[32*64] 25 fc2_W[64*32] 26 fc3_W[32] 27 fc3_b[1] 28 att_W[32*32]
// 29 ntn.Wt[32*32*16] 30 ntn.Wb[16*64] 31 ntn.bt[16]
// 32 ff1_W[16*16] 33 ff1_b[16] 34 ff2_W[16*8] 35 ff2_b[8]
// 36 ff3_W[8*4]  37 ff3_b[4]  38 sc_W[4] 39 sc_b[1]
// 40 edge_index_1[2*E] (int32)  41 edge_index_2[2*E] (int32)
// ----------------------------------------------------------------------------

#define GN 1536
#define GE 24576
#define IC 8   // i values per wave in the energy kernel

typedef __attribute__((ext_vector_type(16))) _Float16 v16h;
typedef __attribute__((ext_vector_type(8)))  float    v8f;

// single-instruction xor-butterfly within 32 lanes (ds_swizzle group-of-32:
// offset = xor_mask<<10 | or_mask<<5 | and_mask; and=0x1f, or=0)
#define SWZ_XOR(x, imm) \
  __int_as_float(__builtin_amdgcn_ds_swizzle(__float_as_int(x), (imm)))

// ---------------------------------------------------------------- utilities
__global__ void k_zero(float* __restrict__ p, int n) {
  int i = blockIdx.x * blockDim.x + threadIdx.x;
  if (i < n) p[i] = 0.f;
}

// agg[dst[e], f] += x[src[e], f]
__global__ void k_scatter_add(const float* __restrict__ x, const int* __restrict__ src,
                              const int* __restrict__ dst, float* __restrict__ agg,
                              int Ee, int d) {
  int idx = blockIdx.x * blockDim.x + threadIdx.x;
  if (idx >= Ee * d) return;
  int e = idx / d, f = idx - e * d;
  atomicAdd(&agg[dst[e] * d + f], x[src[e] * d + f]);
}

// y[n,o] = relu( sum_k ((1+eps)x[n,k] + agg[n,k]) * W[k,o] + b[o] )
__global__ void k_gin_lin0(const float* __restrict__ x, const float* __restrict__ agg,
                           const float* __restrict__ eps, const float* __restrict__ W,
                           const float* __restrict__ b, float* __restrict__ y,
                           int Nn, int din, int dh) {
  int idx = blockIdx.x * blockDim.x + threadIdx.x;
  if (idx >= Nn * dh) return;
  int n = idx / dh, o = idx - n * dh;
  float ep = 1.0f + eps[0];
  const float* xr = x + n * din;
  const float* ar = agg + n * din;
  float acc = b[o];
  for (int k = 0; k < din; ++k) acc += (ep * xr[k] + ar[k]) * W[k * dh + o];
  y[idx] = fmaxf(acc, 0.f);
}

// y[n,o] = sum_k x[n,k]*W[k,o] (+ b[o])   (no relu; b may be null)
__global__ void k_lin(const float* __restrict__ x, const float* __restrict__ W,
                      const float* __restrict__ b, float* __restrict__ y,
                      int Nn, int din, int dh) {
  int idx = blockIdx.x * blockDim.x + threadIdx.x;
  if (idx >= Nn * dh) return;
  int n = idx / dh, o = idx - n * dh;
  const float* xr = x + n * din;
  float acc = b ? b[o] : 0.f;
  for (int k = 0; k < din; ++k) acc += xr[k] * W[k * dh + o];
  y[idx] = acc;
}

// biased batch stats per feature (one block per feature)
__global__ void k_bn_stats(const float* __restrict__ y, float* __restrict__ mean,
                           float* __restrict__ var, int Nn, int d) {
  int f = blockIdx.x, tid = threadIdx.x;
  __shared__ float s1[256], s2[256];
  float a = 0.f, b = 0.f;
  for (int n = tid; n < Nn; n += 256) { float v = y[n * d + f]; a += v; b += v * v; }
  s1[tid] = a; s2[tid] = b; __syncthreads();
  for (int s = 128; s > 0; s >>= 1) {
    if (tid < s) { s1[tid] += s1[tid + s]; s2[tid] += s2[tid + s]; }
    __syncthreads();
  }
  if (tid == 0) { float mu = s1[0] / Nn; mean[f] = mu; var[f] = s2[0] / Nn - mu * mu; }
}

__global__ void k_bn_apply(const float* __restrict__ y, const float* __restrict__ mean,
                           const float* __restrict__ var, const float* __restrict__ gamma,
                           const float* __restrict__ beta, float* __restrict__ out,
                           int Nn, int d, int relu) {
  int idx = blockIdx.x * blockDim.x + threadIdx.x;
  if (idx >= Nn * d) return;
  int f = idx % d;
  float v = (y[idx] - mean[f]) * rsqrtf(var[f] + 1e-5f) * gamma[f] + beta[f];
  out[idx] = relu ? fmaxf(v, 0.f) : v;
}

// -------------------------------------------------- WMMA B-operand packing
// Pack fc2_W [64,32] into 4 groups (kstep s in {0,1}) x (nhalf h in {0,1}).
// B 32x16 f16 layout: lanes 0-15 hold K in {0..7,16..23}, lanes 16-31 hold
// K in {8..15,24..31}; column N = lane&15.
__global__ void k_pack_b(const float* __restrict__ W2, __half* __restrict__ Bp) {
  int tid = blockIdx.x * blockDim.x + threadIdx.x;  // 0..127
  if (tid >= 128) return;
  int lane = tid & 31;
  int g = tid >> 5;               // group = s*2 + h
  int s = g >> 1, h = g & 1;
  int hi = lane >> 4, n16 = lane & 15;
  for (int q = 0; q < 16; ++q) {
    int k = 32 * s + 8 * hi + q + ((q >= 8) ? 8 : 0);
    int n = 16 * h + n16;
    Bp[tid * 16 + q] = __float2half(W2[k * 32 + n]);
  }
}

// ------------------------------------------------------------ energy (WMMA)
// One wave per (16-wide j tile, 8-long i chunk). Per i: A operand row m=j is
// relu(A1[i,:]+A2[j,:]) in f16 (A2 slice register-resident across the chunk);
// B operand = prepacked W2 (register-resident); 2 K-steps x 2 N-halves of
// v_wmma_f32_16x16x32_f16; then relu + dot with fc3_W + stage-major
// ds_swizzle reduce + one predicated store.
__global__ __launch_bounds__(256) void k_energy(
    const float* __restrict__ A1, const float* __restrict__ A2,
    const float* __restrict__ w3, const float* __restrict__ b3,
    const __half* __restrict__ Bpack, float* __restrict__ energy, int Nn) {
  const int lane = threadIdx.x;                     // 0..31
  const int wave = threadIdx.y;                     // 0..7
  const int i0   = blockIdx.y * IC;
  const int j0   = (blockIdx.x * 8 + wave) * 16;
  const int hi   = lane >> 4;
  const int m    = lane & 15;
  const int j    = j0 + m;

  // A 16x32 f16 layout: lanes 0-15 hold K in {0..7,16..23}, lanes 16-31 hold
  // K in {8..15,24..31}; slot order = 2 per VGPR low-half first.
  // Preload this lane's A2 slice for both K-steps (invariant over i).
  float a2v[2 * 16];
  const float* a2 = A2 + j * 64;
#pragma unroll
  for (int s = 0; s < 2; ++s)
#pragma unroll
    for (int t = 0; t < 16; ++t) {
      int k = 32 * s + t + 8 * hi + ((t >= 8) ? 8 : 0);
      a2v[s * 16 + t] = a2[k];
    }

  // B operand tiles (invariant over i and j)
  const v16h* Bp = (const v16h*)Bpack;
  v16h B00 = Bp[0 * 32 + lane];   // kstep 0, nhalf 0
  v16h B01 = Bp[1 * 32 + lane];   // kstep 0, nhalf 1
  v16h B10 = Bp[2 * 32 + lane];   // kstep 1, nhalf 0
  v16h B11 = Bp[3 * 32 + lane];   // kstep 1, nhalf 1

  const float w3a = w3[m];
  const float w3b = w3[m + 16];
  const float bb  = b3[0];

#pragma unroll 1
  for (int ic = 0; ic < IC; ++ic) {
    const int i = i0 + ic;
    const float* a1 = A1 + i * 64;

    v8f acc0 = {}; v8f acc1 = {};
#pragma unroll
    for (int s = 0; s < 2; ++s) {
      v16h a;
#pragma unroll
      for (int t = 0; t < 16; ++t) {
        int k = 32 * s + t + 8 * hi + ((t >= 8) ? 8 : 0);
        float v = a1[k] + a2v[s * 16 + t];
        v = v > 0.f ? v : 0.f;                      // relu of fc1 stage
        a[t] = (_Float16)v;
      }
      if (s == 0) {
        acc0 = __builtin_amdgcn_wmma_f32_16x16x32_f16(false, a, false, B00,
                                                      (short)0, acc0, false, false);
        acc1 = __builtin_amdgcn_wmma_f32_16x16x32_f16(false, a, false, B01,
                                                      (short)0, acc1, false, false);
      } else {
        acc0 = __builtin_amdgcn_wmma_f32_16x16x32_f16(false, a, false, B10,
                                                      (short)0, acc0, false, false);
        acc1 = __builtin_amdgcn_wmma_f32_16x16x32_f16(false, a, false, B11,
                                                      (short)0, acc1, false, false);
      }
    }

    // C/D layout: vgpr v holds row M = 8*hi + v, column N = lane&15.
    // Epilogue: relu + fc3 dot per row, then stage-major xor-butterflies so 8
    // independent ds_swizzles are in flight per stage.
    float p[8], q[8];
#pragma unroll
    for (int v = 0; v < 8; ++v) {
      float x0 = acc0[v] > 0.f ? acc0[v] : 0.f;
      float x1 = acc1[v] > 0.f ? acc1[v] : 0.f;
      p[v] = x0 * w3a + x1 * w3b;
    }
#pragma unroll
    for (int v = 0; v < 8; ++v) q[v] = SWZ_XOR(p[v], 0x041f);   // xor 1
#pragma unroll
    for (int v = 0; v < 8; ++v) p[v] += q[v];
#pragma unroll
    for (int v = 0; v < 8; ++v) q[v] = SWZ_XOR(p[v], 0x081f);   // xor 2
#pragma unroll
    for (int v = 0; v < 8; ++v) p[v] += q[v];
#pragma unroll
    for (int v = 0; v < 8; ++v) q[v] = SWZ_XOR(p[v], 0x101f);   // xor 4
#pragma unroll
    for (int v = 0; v < 8; ++v) p[v] += q[v];
#pragma unroll
    for (int v = 0; v < 8; ++v) q[v] = SWZ_XOR(p[v], 0x201f);   // xor 8
#pragma unroll
    for (int v = 0; v < 8; ++v) p[v] += q[v];
    // every lane now holds all 8 row sums of its half; lane m<8 stores row
    // hi*8+m via a cndmask select tree + one predicated store.
    float r = p[0];
#pragma unroll
    for (int v = 1; v < 8; ++v) r = (m == v) ? p[v] : r;
    if (m < 8) energy[(size_t)i * Nn + (j0 + hi * 8 + m)] = r + bb;
  }
}

// --------------------------------------- row softmax x dots -> cost scalar
__global__ __launch_bounds__(256) void k_row_cost(
    const float* __restrict__ energy, const float* __restrict__ af1,
    const float* __restrict__ af2, float* __restrict__ cost, int Nn) {
  int i = blockIdx.x, tid = threadIdx.x;
  __shared__ float a2s[32];
  __shared__ float red[256], red2[256];
  if (tid < 32) a2s[tid] = af2[i * 32 + tid];
  __syncthreads();
  const float* er = energy + (size_t)i * Nn;
  float mx = -3.402823466e38f;
  for (int j = tid; j < Nn; j += 256) mx = fmaxf(mx, er[j]);
  red[tid] = mx; __syncthreads();
  for (int s = 128; s > 0; s >>= 1) {
    if (tid < s) red[tid] = fmaxf(red[tid], red[tid + s]);
    __syncthreads();
  }
  float rmax = red[0];
  __syncthreads();
  float S = 0.f, T = 0.f;
  for (int j = tid; j < Nn; j += 256) {
    float e = expf(er[j] - rmax);
    const float* a1 = af1 + j * 32;
    float dt = 0.f;
#pragma unroll
    for (int k = 0; k < 32; ++k) dt += a1[k] * a2s[k];   // dots[i,j]
    S += e; T += e * dt;
  }
  red[tid] = S; red2[tid] = T; __syncthreads();
  for (int s = 128; s > 0; s >>= 1) {
    if (tid < s) { red[tid] += red[tid + s]; red2[tid] += red2[tid + s]; }
    __syncthreads();
  }
  if (tid == 0) atomicAdd(cost, red2[0] / red[0]);
}

// ------------------------------------------------------- attention pooling
__global__ void k_colmean(const float* __restrict__ af, float* __restrict__ cm, int Nn) {
  int f = blockIdx.x, tid = threadIdx.x;            // 32 blocks x 256
  __shared__ float s1[256];
  float a = 0.f;
  for (int n = tid; n < Nn; n += 256) a += af[n * 32 + f];
  s1[tid] = a; __syncthreads();
  for (int s = 128; s > 0; s >>= 1) { if (tid < s) s1[tid] += s1[tid + s]; __syncthreads(); }
  if (tid == 0) cm[f] = s1[0] / Nn;
}

__global__ void k_ctx(const float* __restrict__ cm, const float* __restrict__ W,
                      float* __restrict__ ctx) {
  int l = threadIdx.x;                              // 32 threads
  float a = 0.f;
  for (int k = 0; k < 32; ++k) a += cm[k] * W[k * 32 + l];
  ctx[l] = tanhf(a);
}

__global__ void k_pool(const float* __restrict__ af, const float* __restrict__ ctx,
                       float* __restrict__ p, int Nn) {
  __shared__ float sp[32];
  int tid = threadIdx.x;
  if (tid < 32) sp[tid] = 0.f;
  __syncthreads();
  int i = blockIdx.x * blockDim.x + tid;
  if (i < Nn) {
    const float* r = af + i * 32;
    float a = 0.f;
#pragma unroll
    for (int k = 0; k < 32; ++k) a += r[k] * ctx[k];
    float s = 1.f / (1.f + expf(-a));
#pragma unroll
    for (int k = 0; k < 32; ++k) atomicAdd(&sp[k], r[k] * s);
  }
  __syncthreads();
  if (tid < 32) atomicAdd(&p[tid], sp[tid]);
}

// ---------------------------------------------------- NTN + FF head (tiny)
__global__ void k_head(const float* __restrict__ p1, const float* __restrict__ p2,
                       const float* __restrict__ Wt, const float* __restrict__ Wb,
                       const float* __restrict__ bt,
                       const float* __restrict__ ff1W, const float* __restrict__ ff1b,
                       const float* __restrict__ ff2W, const float* __restrict__ ff2b,
                       const float* __restrict__ ff3W, const float* __restrict__ ff3b,
                       const float* __restrict__ scW, const float* __restrict__ scb,
                       const float* __restrict__ cost, const float* __restrict__ hb,
                       float* __restrict__ out) {
  __shared__ float e1[32], e2[32], g[16], h1[16], h2[8], h3[4];
  int tid = threadIdx.x;                            // 64 threads
  if (tid < 32) { e1[tid] = p1[tid]; e2[tid] = p2[tid]; }
  __syncthreads();
  if (tid < 16) {
    float s = 0.f;
    for (int k = 0; k < 32; ++k) {
      float acc = 0.f;
      for (int l = 0; l < 32; ++l) acc += Wt[(k * 32 + l) * 16 + tid] * e2[l];
      s += e1[k] * acc;
    }
    float blk = 0.f;
    for (int c = 0; c < 32; ++c) blk += Wb[tid * 64 + c] * e1[c];
    for (int c = 0; c < 32; ++c) blk += Wb[tid * 64 + 32 + c] * e2[c];
    g[tid] = fmaxf(s + blk + bt[tid], 0.f);
  }
  __syncthreads();
  if (tid < 16) {
    float a = ff1b[tid];
    for (int u = 0; u < 16; ++u) a += g[u] * ff1W[u * 16 + tid];
    h1[tid] = fmaxf(a, 0.f);
  }
  __syncthreads();
  if (tid < 8) {
    float a = ff2b[tid];
    for (int u = 0; u < 16; ++u) a += h1[u] * ff2W[u * 8 + tid];
    h2[tid] = fmaxf(a, 0.f);
  }
  __syncthreads();
  if (tid < 4) {
    float a = ff3b[tid];
    for (int u = 0; u < 8; ++u) a += h2[u] * ff3W[u * 4 + tid];
    h3[tid] = fmaxf(a, 0.f);
  }
  __syncthreads();
  if (tid == 0) {
    float bias = scb[0];
    for (int u = 0; u < 4; ++u) bias += h3[u] * scW[u];
    float score = 1.f / (1.f + expf(-(cost[0] + bias)));
    out[0] = score * hb[0];   // pre_ged
    out[1] = score;           // score
  }
}

// ---------------------------------------------------------------- launcher
extern "C" void kernel_launch(void* const* d_in, const int* in_sizes, int n_in,
                              void* d_out, int out_size, void* d_ws, size_t ws_size,
                              hipStream_t stream) {
  const int Nn = GN, Ee = GE;
  const float* feats1 = (const float*)d_in[0];
  const float* feats2 = (const float*)d_in[1];
  const float* hb     = (const float*)d_in[2];
  const float* eps[3] = {(const float*)d_in[3], (const float*)d_in[4], (const float*)d_in[5]};
  const float *W0[3], *b0[3], *W1[3], *b1[3], *gam[3], *bet[3];
  for (int L = 0; L < 3; ++L) {
    int base = 6 + 6 * L;
    W0[L]  = (const float*)d_in[base + 0];
    b0[L]  = (const float*)d_in[base + 1];
    W1[L]  = (const float*)d_in[base + 2];
    b1[L]  = (const float*)d_in[base + 3];
    gam[L] = (const float*)d_in[base + 4];
    bet[L] = (const float*)d_in[base + 5];
  }
  const float* fc1W = (const float*)d_in[24];
  const float* fc2W = (const float*)d_in[25];
  const float* fc3W = (const float*)d_in[26];
  const float* fc3b = (const float*)d_in[27];
  const float* attW = (const float*)d_in[28];
  const float* Wt   = (const float*)d_in[29];
  const float* Wb   = (const float*)d_in[30];
  const float* bt   = (const float*)d_in[31];
  const float* ff1W = (const float*)d_in[32];
  const float* ff1b = (const float*)d_in[33];
  const float* ff2W = (const float*)d_in[34];
  const float* ff2b = (const float*)d_in[35];
  const float* ff3W = (const float*)d_in[36];
  const float* ff3b = (const float*)d_in[37];
  const float* scW  = (const float*)d_in[38];
  const float* scb  = (const float*)d_in[39];
  const int* edge1  = (const int*)d_in[40];
  const int* edge2  = (const int*)d_in[41];

  // workspace layout (float offsets; all multiples of 8 -> 32B aligned)
  float* ws = (float*)d_ws;
  float* agg    = ws;                       // N*128
  float* t0     = agg    + Nn * 128;        // N*128
  float* t1     = t0     + Nn * 128;        // N*128
  float* xcur   = t1     + Nn * 128;        // N*128
  float* af1    = xcur   + Nn * 128;        // N*32
  float* af2    = af1    + Nn * 32;         // N*32
  float* A1     = af2    + Nn * 32;         // N*64
  float* A2     = A1     + Nn * 64;         // N*64
  float* energy = A2     + Nn * 64;         // N*N
  float* bpackf = energy + (size_t)Nn * Nn; // 1024 floats = 2048 halfs
  __half* Bpack = (__half*)bpackf;
  float* mean   = bpackf + 1024;            // 128
  float* var    = mean   + 128;             // 128
  float* misc   = var    + 128;             // cost + p1 + p2 + cm + ctx
  float* cost   = misc + 0;
  float* p1     = misc + 8;
  float* p2     = misc + 40;
  float* cm     = misc + 72;
  float* ctx    = misc + 104;

  const int din_a[3] = {32, 128, 64};
  const int dh_a[3]  = {128, 64, 32};

  auto conv_pass = [&](const float* feats, const int* edge, float* af) {
    const float* xin = feats;
    for (int L = 0; L < 3; ++L) {
      int din = din_a[L], dh = dh_a[L];
      k_zero<<<(Nn * din + 255) / 256, 256, 0, stream>>>(agg, Nn * din);
      k_scatter_add<<<(Ee * din + 255) / 256, 256, 0, stream>>>(
          xin, edge, edge + Ee, agg, Ee, din);
      k_gin_lin0<<<(Nn * dh + 255) / 256, 256, 0, stream>>>(
          xin, agg, eps[L], W0[L], b0[L], t0, Nn, din, dh);
      k_lin<<<(Nn * dh + 255) / 256, 256, 0, stream>>>(
          t0, W1[L], b1[L], t1, Nn, dh, dh);
      k_bn_stats<<<dh, 256, 0, stream>>>(t1, mean, var, Nn, dh);
      float* outp = (L == 2) ? af : xcur;
      k_bn_apply<<<(Nn * dh + 255) / 256, 256, 0, stream>>>(
          t1, mean, var, gam[L], bet[L], outp, Nn, dh, (L < 2) ? 1 : 0);
      xin = xcur;
    }
  };

  conv_pass(feats1, edge1, af1);
  conv_pass(feats2, edge2, af2);

  // stage-1 projections (fc1 has no bias; relu deferred to pair sum)
  k_lin<<<(Nn * 64 + 255) / 256, 256, 0, stream>>>(af1, fc1W, nullptr, A1, Nn, 32, 64);
  k_lin<<<(Nn * 64 + 255) / 256, 256, 0, stream>>>(af2, fc1W, nullptr, A2, Nn, 32, 64);

  // WMMA pairwise energy: grid = (j-tiles/8, i-chunks)
  k_pack_b<<<1, 128, 0, stream>>>(fc2W, Bpack);
  k_energy<<<dim3(Nn / 128, Nn / IC), dim3(32, 8), 0, stream>>>(
      A1, A2, fc3W, fc3b, Bpack, energy, Nn);

  // zero accumulators (cost, p1, p2)
  k_zero<<<1, 256, 0, stream>>>(misc, 136);

  // cost_sum = sum_i sum_j softmax(energy)_ij * <af1[j], af2[i]>
  k_row_cost<<<Nn, 256, 0, stream>>>(energy, af1, af2, cost, Nn);

  // attention pooling for both graphs (mean commutes with the linear)
  k_colmean<<<32, 256, 0, stream>>>(af1, cm, Nn);
  k_ctx<<<1, 32, 0, stream>>>(cm, attW, ctx);
  k_pool<<<(Nn + 255) / 256, 256, 0, stream>>>(af1, ctx, p1, Nn);

  k_colmean<<<32, 256, 0, stream>>>(af2, cm, Nn);
  k_ctx<<<1, 32, 0, stream>>>(cm, attW, ctx);
  k_pool<<<(Nn + 255) / 256, 256, 0, stream>>>(af2, ctx, p2, Nn);

  // NTN + feed-forward head -> [pre_ged, score]
  k_head<<<1, 64, 0, stream>>>(p1, p2, Wt, Wb, bt, ff1W, ff1b, ff2W, ff2b,
                               ff3W, ff3b, scW, scb, cost, hb, (float*)d_out);
}
